// TemporalDecoder_73761768341812
// MI455X (gfx1250) — matrix operand order repaired and checked by reading, not compiled, not run on hardware
//
#include <hip/hip_runtime.h>

// ---------------------------------------------------------------------------
// TemporalDecoder for MI455X (gfx1250, wave32, WMMA bf16 16x16x32)
//
// Math reductions vs reference:
//  * softmax over seq axis cancels the hid-dependent and bias terms ->
//    attention context is recurrence-independent, precomputed once.
//  * OUT==1 -> x feedback is rank-1: gx = pred*u + c  (u=W_ih@proj_W, c=W_ih@proj_b+b_ih)
// Remaining GEMMs (enc@attnW2^T once; hid@W_hh^T per step) run on
// v_wmma_f32_16x16x32_bf16 with f32 accumulation.
// ---------------------------------------------------------------------------

typedef __attribute__((ext_vector_type(16))) __bf16 v16bf;
typedef __attribute__((ext_vector_type(8)))  float  v8f;

struct U4 { unsigned int x, y, z, w; };     // 16B, trivially copyable
struct Frag32 { U4 lo, hi; };               // 32B == v16bf

static __device__ __forceinline__ unsigned short f2bf(float f) {
  __bf16 b = (__bf16)f;                     // RNE convert
  return __builtin_bit_cast(unsigned short, b);
}
static __device__ __forceinline__ float bf2f(unsigned short u) {
  __bf16 b = __builtin_bit_cast(__bf16, u);
  return (float)b;
}

#define BN_TOTAL 16000
#define H 128
#define S 48
#define T 12
#define ENC_STR 136   // padded ushort stride for enc tile in LDS (bank spread)
#define HID_STR 132   // padded float stride for hid in LDS (bank spread)

// ---------------------------------------------------------------------------
// Kernel 1: tiny prep — u, c vectors and bf16 weight copies
// ---------------------------------------------------------------------------
__global__ __launch_bounds__(256) void prep_kernel(
    const float* __restrict__ W_ih, const float* __restrict__ b_ih,
    const float* __restrict__ W_hh, const float* __restrict__ attn_W,
    const float* __restrict__ proj_W, const float* __restrict__ proj_b,
    float* __restrict__ u, float* __restrict__ c,
    unsigned short* __restrict__ whhb, unsigned short* __restrict__ attw2b) {
  const int gid  = blockIdx.x * blockDim.x + threadIdx.x;
  const int nthr = gridDim.x * blockDim.x;
  for (int j = gid; j < 3 * H; j += nthr) {       // 384 rows
    float su = 0.f, sc = 0.f;
    for (int i = 0; i < H; ++i) {
      float w = W_ih[j * H + i];
      su = fmaf(w, proj_W[i], su);                // proj_W: [128,1]
      sc = fmaf(w, proj_b[i], sc);
    }
    u[j] = su;
    c[j] = sc + b_ih[j];
  }
  for (int i = gid; i < 3 * H * H; i += nthr) whhb[i] = f2bf(W_hh[i]);
  for (int i = gid; i < H * H; i += nthr) {
    int hh = i >> 7, k = i & 127;
    attw2b[i] = f2bf(attn_W[hh * 256 + 128 + k]); // attn_W[:, H:2H]
  }
}

// ---------------------------------------------------------------------------
// Kernel 2: context precompute. One block (4 waves) per bn.
//   E = enc @ attnW2^T  (WMMA bf16) ; w = softmax_s(E) ; ctx = sum_s w*enc
// ---------------------------------------------------------------------------
__global__ __launch_bounds__(128) void ctx_kernel(
    const float* __restrict__ enc, const unsigned short* __restrict__ attw2b,
    float* __restrict__ ctxw) {
  __shared__ unsigned short encb[S * ENC_STR];
  __shared__ float E[S * H];

  const int bn  = blockIdx.x;
  const int tid = threadIdx.x;
  const float* eg = enc + (size_t)bn * (S * H);

  for (int i = tid; i < S * H; i += 128) {
    int r = i >> 7, k = i & 127;
    encb[r * ENC_STR + k] = f2bf(eg[i]);
  }
  __syncthreads();

  const int wave = tid >> 5, lane = tid & 31;
  const int l15 = lane & 15;
  const int aoff = (lane & 16) ? 8 : 0;    // A-matrix K interleave
  const int boff = (lane & 16) ? 8 : 0;    // B-matrix K interleave

  for (int j = 0; j < 6; ++j) {            // 24 tiles / 4 waves
    int ti = wave * 6 + j;
    int mt = ti >> 3, nt = ti & 7;
    v8f acc = {0.f, 0.f, 0.f, 0.f, 0.f, 0.f, 0.f, 0.f};
    int arow = mt * 16 + l15;
    int brow = nt * 16 + l15;
#pragma unroll
    for (int kc = 0; kc < 4; ++kc) {
      int ka = kc * 32 + aoff;
      int kb = kc * 32 + boff;
      Frag32 fa, fb;
      fa.lo = *(const U4*)&encb[arow * ENC_STR + ka];
      fa.hi = *(const U4*)&encb[arow * ENC_STR + ka + 16];
      fb.lo = *(const U4*)&attw2b[brow * H + kb];
      fb.hi = *(const U4*)&attw2b[brow * H + kb + 16];
      v16bf a = __builtin_bit_cast(v16bf, fa);
      v16bf b = __builtin_bit_cast(v16bf, fb);
      acc = __builtin_amdgcn_wmma_f32_16x16x32_bf16(false, a, false, b,
                                                    (short)0, acc, false, false);
    }
    int col   = nt * 16 + l15;
    int rbase = mt * 16 + ((lane & 16) ? 8 : 0);
#pragma unroll
    for (int i = 0; i < 8; ++i) E[(rbase + i) * H + col] = acc[i];
  }
  __syncthreads();

  // per-feature softmax over s, then weighted sum of enc
  const int h = tid;                        // 0..127
  float mx = -3.0e38f;
  for (int s = 0; s < S; ++s) mx = fmaxf(mx, E[s * H + h]);
  float sum = 0.f;
  for (int s = 0; s < S; ++s) {
    float e = __expf(E[s * H + h] - mx);
    E[s * H + h] = e;
    sum += e;
  }
  float cx = 0.f;
  for (int s = 0; s < S; ++s) cx = fmaf(E[s * H + h], eg[s * H + h], cx);
  ctxw[(size_t)bn * H + h] = cx / sum;
}

// ---------------------------------------------------------------------------
// Kernel 3: GRU recurrence, 32 rows per block, 8 waves, 12 steps in-block.
//   per step: gh = hid @ W_hh^T via WMMA -> LDS(bf16); fused gates; pred
// ---------------------------------------------------------------------------
__global__ __launch_bounds__(256) void gru_kernel(
    const float* __restrict__ enc_state, const float* __restrict__ b_ih,
    const float* __restrict__ b_hh, const float* __restrict__ fc_W,
    const float* __restrict__ fc_b, const float* __restrict__ u,
    const float* __restrict__ c, const unsigned short* __restrict__ whhb,
    const float* __restrict__ ctxw, float* __restrict__ out) {
  __shared__ float hidL[32 * HID_STR];
  __shared__ unsigned short ghb[32 * 384];
  __shared__ float uL[384], cL[384], bihL[384], bhhL[384], fcL[128];
  __shared__ float predL[32], predA[32];

  const int tid = threadIdx.x;
  const int r0  = blockIdx.x * 32;

  for (int i = tid; i < 32 * H; i += 256) {
    int r = i >> 7, k = i & 127;
    hidL[r * HID_STR + k] = enc_state[(size_t)r0 * H + i];
  }
  for (int i = tid; i < 384; i += 256) {
    uL[i] = u[i]; cL[i] = c[i]; bihL[i] = b_ih[i]; bhhL[i] = b_hh[i];
  }
  if (tid < 128) fcL[tid] = fc_W[tid];
  if (tid < 32)  { predL[tid] = 0.f; predA[tid] = 0.f; }
  __syncthreads();

  // ---- per-thread constants for the elementwise phase (h fixed per thread)
  const int h = tid & 127;
  const int rpar = tid >> 7;                 // row parity
  const float u_r = uL[h], u_z = uL[h + 128], u_n = uL[h + 256];
  const float c_r = cL[h], c_z = cL[h + 128], c_n = cL[h + 256];
  const float bi_r = bihL[h], bi_z = bihL[h + 128], bi_n = bihL[h + 256];
  const float bh_r = bhhL[h], bh_z = bhhL[h + 128], bh_n = bhhL[h + 256];
  const float fcw = fcL[h];
  const float fcb = fc_b[0];
  float hreg[16], ctxr[16];
#pragma unroll
  for (int i = 0; i < 16; ++i) {
    int row = 2 * i + rpar;
    hreg[i] = hidL[row * HID_STR + h];
    ctxr[i] = ctxw[(size_t)(r0 + row) * H + h];
  }

  // ---- WMMA assignment: 48 tiles (2 M x 24 N), 6 N-tiles per wave
  const int wave = tid >> 5, lane = tid & 31;
  const int l15 = lane & 15;
  const int aoff = (lane & 16) ? 8 : 0;
  const int boff = (lane & 16) ? 8 : 0;
  const int m = wave & 1;
  const int nbase = wave >> 1;               // 0..3
  const int arow = m * 16 + l15;
  const int rbase = m * 16 + ((lane & 16) ? 8 : 0);

  for (int t = 0; t < T; ++t) {
    // ---------------- Phase W: gh = hid @ W_hh^T ----------------
    v16bf afr[4];
#pragma unroll
    for (int kc = 0; kc < 4; ++kc) {
      int ka = kc * 32 + aoff;
      v16bf a;
#pragma unroll
      for (int q = 0; q < 8; ++q) {
        a[q]     = (__bf16)hidL[arow * HID_STR + ka + q];
        a[q + 8] = (__bf16)hidL[arow * HID_STR + ka + 16 + q];
      }
      afr[kc] = a;
    }
#pragma unroll
    for (int j = 0; j < 6; ++j) {
      int n = nbase + 4 * j;                 // 0..23
      int brow = n * 16 + l15;
      v8f acc = {0.f, 0.f, 0.f, 0.f, 0.f, 0.f, 0.f, 0.f};
#pragma unroll
      for (int kc = 0; kc < 4; ++kc) {
        int kb = kc * 32 + boff;
        Frag32 fb;
        fb.lo = *(const U4*)&whhb[brow * H + kb];
        fb.hi = *(const U4*)&whhb[brow * H + kb + 16];
        v16bf b = __builtin_bit_cast(v16bf, fb);
        acc = __builtin_amdgcn_wmma_f32_16x16x32_bf16(false, afr[kc], false, b,
                                                      (short)0, acc, false, false);
      }
      int col = n * 16 + l15;
#pragma unroll
      for (int i = 0; i < 8; ++i) ghb[(rbase + i) * 384 + col] = f2bf(acc[i]);
    }
    __syncthreads();

    // ---------------- Phase E: gates + pred reduction ----------------
#pragma unroll
    for (int i = 0; i < 16; ++i) {
      int row = 2 * i + rpar;
      float gh_r = bf2f(ghb[row * 384 + h]);
      float gh_z = bf2f(ghb[row * 384 + 128 + h]);
      float gh_n = bf2f(ghb[row * 384 + 256 + h]);
      float p = predL[row];
      float gx_r, gx_z, gx_n;
      if (t == 0) { gx_r = bi_r; gx_z = bi_z; gx_n = bi_n; }  // x0 == 0
      else {
        gx_r = fmaf(p, u_r, c_r);
        gx_z = fmaf(p, u_z, c_z);
        gx_n = fmaf(p, u_n, c_n);
      }
      float rg = 1.f / (1.f + __expf(-(gx_r + gh_r + bh_r)));
      float zg = 1.f / (1.f + __expf(-(gx_z + gh_z + bh_z)));
      float nn = tanhf(gx_n + rg * (gh_n + bh_n));
      float hn = (1.f - zg) * nn + zg * hreg[i];
      hreg[i] = hn;
      hidL[row * HID_STR + h] = hn;
      float val = fcw * (hn + ctxr[i]);      // partial of pred = fc_W . (hid+ctx)
#pragma unroll
      for (int off = 16; off >= 1; off >>= 1) val += __shfl_down(val, off, 32);
      if (lane == 0) atomicAdd(&predA[row], val);   // ds_add_f32
    }
    __syncthreads();
    if (tid < 32) {
      float pv = predA[tid] + fcb;
      predL[tid] = pv;
      predA[tid] = 0.f;
      out[(size_t)(r0 + tid) * T + t] = pv;  // [BN, T, 1] layout
    }
    __syncthreads();
  }
}

// ---------------------------------------------------------------------------
extern "C" void kernel_launch(void* const* d_in, const int* in_sizes, int n_in,
                              void* d_out, int out_size, void* d_ws, size_t ws_size,
                              hipStream_t stream) {
  (void)in_sizes; (void)n_in; (void)out_size; (void)ws_size;
  const float* enc_state = (const float*)d_in[0];   // [B,N,H]
  const float* enc_out   = (const float*)d_in[1];   // [B,N,S,H]
  const float* W_ih      = (const float*)d_in[2];   // [3H,IN]
  const float* W_hh      = (const float*)d_in[3];   // [3H,H]
  const float* b_ih      = (const float*)d_in[4];
  const float* b_hh      = (const float*)d_in[5];
  const float* attn_W    = (const float*)d_in[6];   // [H,2H]
  // d_in[7] = attn_b : cancels in softmax, unused
  const float* fc_W      = (const float*)d_in[8];   // [1,H]
  const float* fc_b      = (const float*)d_in[9];   // [1]
  const float* proj_W    = (const float*)d_in[10];  // [IN,1]
  const float* proj_b    = (const float*)d_in[11];  // [IN]
  float* out = (float*)d_out;

  char* ws = (char*)d_ws;
  float* ctxw = (float*)ws;                                   // 16000*128 f32 = 8,192,000 B
  float* u    = (float*)(ws + 8192000);                       // 384 f32
  float* c    = u + 384;                                      // 384 f32
  unsigned short* whhb   = (unsigned short*)(ws + 8192000 + 3072);  // 384*128 bf16
  unsigned short* attw2b = whhb + 384 * 128;                  // 128*128 bf16

  prep_kernel<<<64, 256, 0, stream>>>(W_ih, b_ih, W_hh, attn_W, proj_W, proj_b,
                                      u, c, whhb, attw2b);
  ctx_kernel<<<BN_TOTAL, 128, 0, stream>>>(enc_out, attw2b, ctxw);
  gru_kernel<<<BN_TOTAL / 32, 256, 0, stream>>>(enc_state, b_ih, b_hh, fc_W, fc_b,
                                                u, c, whhb, ctxw, out);
}
